// TemplateDeformNet_35330400977027
// MI455X (gfx1250) — matrix operand order repaired and checked by reading, not compiled, not run on hardware
//
#include <hip/hip_runtime.h>
#include <hip/hip_bf16.h>
#include <stdint.h>

typedef __attribute__((ext_vector_type(8)))  _Float16 v8h;
typedef __attribute__((ext_vector_type(16))) _Float16 v16h;
typedef __attribute__((ext_vector_type(8)))  float    v8f;

#define B_   4
#define T_   16384
#define S_   8192
#define D_   256
#define G_   512
#define INP_ 771
#define KP1_ 896          // 771 padded to multiple of KC_
#define KC_  128          // K-chunk staged in LDS per double-buffer slot
#define M_   (B_*T_)      // 65536 template points total

// ---------------------------------------------------------------------------
// CDNA5 async global->LDS copy (per-lane 16B), tracked by ASYNCcnt.
// ---------------------------------------------------------------------------
__device__ __forceinline__ void async_b128(void* lds_dst, const void* gsrc) {
  uint32_t loff = (uint32_t)(uintptr_t)lds_dst;
  asm volatile("global_load_async_to_lds_b128 %0, %1, off"
               :: "v"(loff), "v"(gsrc) : "memory");
}

// ---------------------------------------------------------------------------
// Weight prep: W (K x N, f32) -> Wt (N x Kpad, f16), zero-padded in K.
// ---------------------------------------------------------------------------
__global__ void wprep_kernel(const float* __restrict__ W, _Float16* __restrict__ Wt,
                             int K, int N, int Kpad) {
  int idx = blockIdx.x * blockDim.x + threadIdx.x;
  if (idx >= N * Kpad) return;
  int n = idx / Kpad, k = idx - n * Kpad;
  Wt[(size_t)n * Kpad + k] = (k < K) ? (_Float16)W[(size_t)k * N + n] : (_Float16)0.f;
}

// ---------------------------------------------------------------------------
// KNN: one thread per template point, surf points tiled through LDS.
// ---------------------------------------------------------------------------
__global__ __launch_bounds__(128) void knn_kernel(
    const float* __restrict__ tmpl, const float* __restrict__ surf,
    int* __restrict__ knn) {
  __shared__ float sx[128], sy[128], sz[128];
  const int tid = threadIdx.x;
  const int i = blockIdx.x * 128 + tid;
  const int b = i / T_;
  const float tx = tmpl[(size_t)i*3+0], ty = tmpl[(size_t)i*3+1], tz = tmpl[(size_t)i*3+2];
  float bd[8]; int bi[8];
  #pragma unroll
  for (int k = 0; k < 8; ++k) { bd[k] = 3.0e38f; bi[k] = 0; }
  const float* sb = surf + (size_t)b * S_ * 3;
  for (int s0 = 0; s0 < S_; s0 += 128) {
    __syncthreads();
    sx[tid] = sb[(size_t)(s0+tid)*3+0];
    sy[tid] = sb[(size_t)(s0+tid)*3+1];
    sz[tid] = sb[(size_t)(s0+tid)*3+2];
    __syncthreads();
    #pragma unroll 4
    for (int j = 0; j < 128; ++j) {
      float dx = tx - sx[j], dy = ty - sy[j], dz = tz - sz[j];
      float d2 = dx*dx + dy*dy + dz*dz;
      if (d2 < bd[7]) {
        int p = 7;
        while (p > 0 && bd[p-1] > d2) { bd[p] = bd[p-1]; bi[p] = bi[p-1]; --p; }
        bd[p] = d2; bi[p] = s0 + j;
      }
    }
  }
  #pragma unroll
  for (int k = 0; k < 8; ++k) knn[(size_t)i*8 + k] = bi[k];
}

// ---------------------------------------------------------------------------
// Build fp16 input matrix X (M x 896): [xyz(3) | knn-mean feats(256) |
// global(512) | zero pad(125)]. One block (256 thr) per template point.
// ---------------------------------------------------------------------------
__global__ __launch_bounds__(256) void build_x_kernel(
    const float* __restrict__ tmpl, const float* __restrict__ gfeat,
    const float* __restrict__ pfeat, const int* __restrict__ knn,
    _Float16* __restrict__ Xh) {
  const int i = blockIdx.x;
  const int b = i / T_;
  const int d = threadIdx.x;        // 0..255
  _Float16* xr = Xh + (size_t)i * KP1_;
  const int* id = knn + (size_t)i * 8;
  const float* pb = pfeat + (size_t)b * S_ * D_;
  float s = 0.f;
  #pragma unroll
  for (int k = 0; k < 8; ++k) s += pb[(size_t)id[k] * D_ + d];
  xr[3 + d] = (_Float16)(s * 0.125f);
  const float* gb = gfeat + (size_t)b * G_;
  xr[259 + d]       = (_Float16)gb[d];
  xr[259 + 256 + d] = (_Float16)gb[256 + d];
  if (d < 3)   xr[d] = (_Float16)tmpl[(size_t)i*3 + d];
  if (d < KP1_ - INP_) xr[INP_ + d] = (_Float16)0.f;   // 125 pad cols
}

// ---------------------------------------------------------------------------
// WMMA GEMM + bias + ReLU:  Out(MxN,f16) = relu(A(MxKpad,f16) @ Bt^T + bias)
// Bt is N x Kpad. Block = 256 thr = 8 waves; block tile = 256(M) x 64(N);
// each wave owns TWO 16-row M-subtiles x four 16-col N-subtiles, so every
// LDS B fragment feeds 2 WMMAs (8 wmma per 8 ds_load_b128 per K-step).
// B tile staged in LDS via async global->LDS (ASYNCcnt), double-buffered.
// Fragment layouts per ISA 7.12.2:
//   A: lane<16 -> row=lane, K=kk+{0..7,16..23}; lane>=16 -> K=kk+{8..15,24..31}
//   B: lane<16 -> col=lane, K=kk+{0..7,16..23}; lane>=16 -> K=kk+{8..15,24..31}
// ---------------------------------------------------------------------------
__global__ __launch_bounds__(256) void gemm_wmma_relu_kernel(
    const _Float16* __restrict__ A, const _Float16* __restrict__ Bt,
    const float* __restrict__ bias, _Float16* __restrict__ Out,
    int N, int Kpad) {
  // +8 halves of row padding => 272B row stride => 4-bank rotation per row.
  __shared__ __align__(16) _Float16 Bs[2][64][KC_ + 8];

  const int tid   = threadIdx.x;
  const int wave  = tid >> 5;
  const int lane  = tid & 31;
  const int lhalf = lane >> 4;
  const int l16   = lane & 15;
  const int row0  = blockIdx.x * 256 + wave * 32;   // 2 x 16-row subtiles
  const int col0  = blockIdx.y * 64;
  const int nchunks = Kpad / KC_;

  // Cooperative async issue of one 64 x KC_ B-chunk: 1024 b128 vectors,
  // 4 per thread, consecutive tids -> consecutive 16B segments of a row.
  auto issue_chunk = [&](int buf, int kk) {
    #pragma unroll
    for (int q = 0; q < 4; ++q) {
      int v  = q * 256 + tid;
      int r  = v >> 4;             // 0..63
      int ko = (v & 15) << 3;      // 0..120, step 8 halves
      async_b128(&Bs[buf][r][ko],
                 Bt + (size_t)(col0 + r) * Kpad + kk + ko);
    }
  };

  v8f acc[2][4];
  {
    v8f zero = {};
    #pragma unroll
    for (int s = 0; s < 2; ++s)
      #pragma unroll
      for (int c = 0; c < 4; ++c) acc[s][c] = zero;
  }

  const _Float16* arow0 = A + (size_t)(row0 + l16) * Kpad + (lhalf << 3);
  const _Float16* arow1 = arow0 + (size_t)16 * Kpad;

  issue_chunk(0, 0);
  for (int chunk = 0; chunk < nchunks; ++chunk) {
    const int buf = chunk & 1;
    if (chunk + 1 < nchunks) {
      issue_chunk(buf ^ 1, (chunk + 1) * KC_);
      asm volatile("s_wait_asynccnt 0x4" ::: "memory");  // current chunk done
    } else {
      asm volatile("s_wait_asynccnt 0x0" ::: "memory");
    }
    __syncthreads();  // publish chunk `buf` to all waves

    const int kk0 = chunk * KC_;
    #pragma unroll
    for (int ks = 0; ks < KC_ / 32; ++ks) {
      // A fragments for both M-subtiles.
      v8h a0lo = *(const v8h*)(arow0 + kk0 + ks*32);
      v8h a0hi = *(const v8h*)(arow0 + kk0 + ks*32 + 16);
      v8h a1lo = *(const v8h*)(arow1 + kk0 + ks*32);
      v8h a1hi = *(const v8h*)(arow1 + kk0 + ks*32 + 16);
      v16h af0 = __builtin_shufflevector(a0lo, a0hi,
          0,1,2,3,4,5,6,7,8,9,10,11,12,13,14,15);
      v16h af1 = __builtin_shufflevector(a1lo, a1hi,
          0,1,2,3,4,5,6,7,8,9,10,11,12,13,14,15);
      // All 4 B fragments up front, then 8 back-to-back WMMAs.
      v16h bf[4];
      #pragma unroll
      for (int c = 0; c < 4; ++c) {
        const _Float16* bp = &Bs[buf][c*16 + l16][ks*32 + (lhalf << 3)];
        v8h blo = *(const v8h*)bp;
        v8h bhi = *(const v8h*)(bp + 16);
        bf[c] = __builtin_shufflevector(blo, bhi,
            0,1,2,3,4,5,6,7,8,9,10,11,12,13,14,15);
      }
      #pragma unroll
      for (int c = 0; c < 4; ++c)
        acc[0][c] = __builtin_amdgcn_wmma_f32_16x16x32_f16(
            false, af0, false, bf[c], (short)0, acc[0][c], false, false);
      #pragma unroll
      for (int c = 0; c < 4; ++c)
        acc[1][c] = __builtin_amdgcn_wmma_f32_16x16x32_f16(
            false, af1, false, bf[c], (short)0, acc[1][c], false, false);
    }
    __syncthreads();  // all waves done with `buf` before DMA overwrites it
  }

  // Epilogue. C/D layout: lane holds col=l16; VGPR r -> row = r + 8*lhalf.
  #pragma unroll
  for (int s = 0; s < 2; ++s) {
    const int rbase = row0 + s*16 + (lhalf << 3);
    #pragma unroll
    for (int c = 0; c < 4; ++c) {
      const int col = col0 + c*16 + l16;
      const float bv = bias[col];
      #pragma unroll
      for (int r = 0; r < 8; ++r) {
        float v = acc[s][c][r] + bv;
        v = v > 0.f ? v : 0.f;
        Out[(size_t)(rbase + r) * N + col] = (_Float16)v;
      }
    }
  }
}

// ---------------------------------------------------------------------------
// Final heads: disp = H2(256)@dw3+db3 ; mat = sigmoid(M2(64)@mw3+mb3).
// One wave per template point, cross-lane reduction.
// ---------------------------------------------------------------------------
__global__ __launch_bounds__(256) void finalize_kernel(
    const _Float16* __restrict__ H2, const _Float16* __restrict__ M2,
    const float* __restrict__ dw3, const float* __restrict__ db3,
    const float* __restrict__ mw3, const float* __restrict__ mb3,
    float* __restrict__ disp, float* __restrict__ mat) {
  const int wave = threadIdx.x >> 5, lane = threadIdx.x & 31;
  const int i = blockIdx.x * 8 + wave;
  const _Float16* h = H2 + (size_t)i * 256;
  float a0 = 0.f, a1 = 0.f, a2 = 0.f;
  #pragma unroll
  for (int k = lane; k < 256; k += 32) {
    float hv = (float)h[k];
    a0 += hv * dw3[k*3+0];
    a1 += hv * dw3[k*3+1];
    a2 += hv * dw3[k*3+2];
  }
  const _Float16* m = M2 + (size_t)i * 64;
  float am = 0.f;
  #pragma unroll
  for (int k = lane; k < 64; k += 32) am += (float)m[k] * mw3[k];
  #pragma unroll
  for (int off = 16; off >= 1; off >>= 1) {
    a0 += __shfl_xor(a0, off, 32);
    a1 += __shfl_xor(a1, off, 32);
    a2 += __shfl_xor(a2, off, 32);
    am += __shfl_xor(am, off, 32);
  }
  if (lane == 0) {
    disp[(size_t)i*3+0] = a0 + db3[0];
    disp[(size_t)i*3+1] = a1 + db3[1];
    disp[(size_t)i*3+2] = a2 + db3[2];
    mat[i] = 1.f / (1.f + __expf(-(am + mb3[0])));
  }
}

// ---------------------------------------------------------------------------
extern "C" void kernel_launch(void* const* d_in, const int* in_sizes, int n_in,
                              void* d_out, int out_size, void* d_ws, size_t ws_size,
                              hipStream_t stream) {
  (void)in_sizes; (void)n_in; (void)out_size; (void)ws_size;
  const float* tmpl  = (const float*)d_in[0];
  const float* surf  = (const float*)d_in[1];
  const float* gfeat = (const float*)d_in[2];
  const float* pfeat = (const float*)d_in[3];
  const float* dw1 = (const float*)d_in[4];
  const float* db1 = (const float*)d_in[5];
  const float* dw2 = (const float*)d_in[6];
  const float* db2 = (const float*)d_in[7];
  const float* dw3 = (const float*)d_in[8];
  const float* db3 = (const float*)d_in[9];
  const float* mw1 = (const float*)d_in[10];
  const float* mb1 = (const float*)d_in[11];
  const float* mw2 = (const float*)d_in[12];
  const float* mb2 = (const float*)d_in[13];
  const float* mw3 = (const float*)d_in[14];
  const float* mb3 = (const float*)d_in[15];

  char* ws = (char*)d_ws;
  size_t off = 0;
  auto alloc = [&](size_t bytes) -> void* {
    void* p = ws + off;
    off = (off + bytes + 255) & ~(size_t)255;
    return p;
  };
  int*      knn  = (int*)     alloc((size_t)M_ * 8 * sizeof(int));
  _Float16* Xh   = (_Float16*)alloc((size_t)M_ * KP1_ * 2);
  _Float16* H1   = (_Float16*)alloc((size_t)M_ * 256 * 2);
  _Float16* H2   = (_Float16*)alloc((size_t)M_ * 256 * 2);
  _Float16* Mm1  = (_Float16*)alloc((size_t)M_ * 128 * 2);
  _Float16* Mm2  = (_Float16*)alloc((size_t)M_ * 64  * 2);
  _Float16* W1t  = (_Float16*)alloc((size_t)256 * KP1_ * 2);
  _Float16* W2t  = (_Float16*)alloc((size_t)256 * 256 * 2);
  _Float16* MW1t = (_Float16*)alloc((size_t)128 * KP1_ * 2);
  _Float16* MW2t = (_Float16*)alloc((size_t)64  * 128 * 2);

  // Weight transpose + fp16 convert + K-pad.
  { int n = 256*KP1_; wprep_kernel<<<(n+255)/256,256,0,stream>>>(dw1, W1t, INP_, 256, KP1_); }
  { int n = 256*256;  wprep_kernel<<<(n+255)/256,256,0,stream>>>(dw2, W2t, 256, 256, 256); }
  { int n = 128*KP1_; wprep_kernel<<<(n+255)/256,256,0,stream>>>(mw1, MW1t, INP_, 128, KP1_); }
  { int n = 64*128;   wprep_kernel<<<(n+255)/256,256,0,stream>>>(mw2, MW2t, 128, 64, 128); }

  // KNN + fused gather-mean / concat / fp16 convert.
  knn_kernel<<<M_/128, 128, 0, stream>>>(tmpl, surf, knn);
  build_x_kernel<<<M_, 256, 0, stream>>>(tmpl, gfeat, pfeat, knn, Xh);

  // WMMA MLP layers (all ReLU), async-LDS staged weights.
  gemm_wmma_relu_kernel<<<dim3(M_/256, 256/64), 256, 0, stream>>>(Xh,  W1t,  db1, H1,  256, KP1_);
  gemm_wmma_relu_kernel<<<dim3(M_/256, 256/64), 256, 0, stream>>>(H1,  W2t,  db2, H2,  256, 256);
  gemm_wmma_relu_kernel<<<dim3(M_/256, 128/64), 256, 0, stream>>>(Xh,  MW1t, mb1, Mm1, 128, KP1_);
  gemm_wmma_relu_kernel<<<dim3(M_/256, 64/64),  256, 0, stream>>>(Mm1, MW2t, mb2, Mm2, 64,  128);

  // Output heads: disp (B*T*3 floats) then mat (B*T floats), concatenated.
  float* disp = (float*)d_out;
  float* mat  = disp + (size_t)M_ * 3;
  finalize_kernel<<<M_/8, 256, 0, stream>>>(H2, Mm2, dw3, db3, mw3, mb3, disp, mat);
}
